// MMAEKNN_42563125903683
// MI455X (gfx1250) — matrix operand order repaired and checked by reading, not compiled, not run on hardware
//
#include <hip/hip_runtime.h>
#include <hip/hip_bf16.h>

// ---------------------------------------------------------------------------
// MI455X (gfx1250): AE-forward + kNN distance loss.
//  - x-Gram (8192x8192x1024, 137 GFLOP) on V_WMMA_F32_16X16X32_BF16 via
//    split-precision bf16x3 (hi/lo planes): ~2e-7 relative d^2 error, far
//    below kNN neighbor gaps, ~2.7x fewer matrix instrs than the f32 path
//    and on the fast 16-bit matrix pipe.
//  - AE GEMMs + z-Gram (K=64) on exact V_WMMA_F32_16X16X4_F32.
// ---------------------------------------------------------------------------

typedef float v2f __attribute__((ext_vector_type(2)));
typedef float v8f __attribute__((ext_vector_type(8)));
typedef __bf16 v8bf __attribute__((ext_vector_type(8)));
typedef __bf16 v16bf __attribute__((ext_vector_type(16)));

#define BB   8192
#define DD   1024
#define HH   512
#define LL   64
#define KNN  15
#define EPSF 1e-8f

static __device__ __forceinline__ v8f wmma_f32_4(v2f a, v2f b, v8f c) {
  return __builtin_amdgcn_wmma_f32_16x16x4_f32(false, a, false, b, (short)0, c,
                                               false, false);
}
static __device__ __forceinline__ v8f wmma_bf16_32(v16bf a, v16bf b, v8f c) {
  return __builtin_amdgcn_wmma_f32_16x16x32_bf16(false, a, false, b, (short)0,
                                                 c, false, false);
}
static __device__ __forceinline__ v16bf cat8(v8bf a, v8bf b) {
  return __builtin_shufflevector(a, b, 0, 1, 2, 3, 4, 5, 6, 7, 8, 9, 10, 11,
                                 12, 13, 14, 15);
}

// ---------------------------------------------------------------------------
// Exact-f32 WMMA GEMM (AE matmuls, z-Gram).  See round-1 notes.
//   TRANSB=0: B[K,N] row-major   TRANSB=1: B[N,K] row-major (Gram)
// EPI: 0 bias+relu  1 bias  2 bias+fused sum((C-Xref)^2)  3 dist epilogue
// 256 thr = 8 waves, wave tile 32x32, block tile 64x128. No LDS/barriers ->
// wave-uniform early exit is safe, EXEC all-ones for every WMMA.
// ---------------------------------------------------------------------------
template <int EPI, int TRANSB>
__global__ __launch_bounds__(256) void gemm_wmma_f32(
    const float* __restrict__ A, const float* __restrict__ Bm,
    const float* __restrict__ aux, const float* __restrict__ Xref,
    float* __restrict__ C, float* __restrict__ accum,
    int M, int N, int K, int lda, int ldb, int ldc) {
  const int lane = threadIdx.x & 31;
  const int wave = threadIdx.x >> 5;
  const int wm = wave & 1;
  const int wn = wave >> 1;
  const int m_base = blockIdx.y * 64 + wm * 32;
  const int n_base = blockIdx.x * 128 + wn * 32;
  if (m_base >= M || n_base >= N) return;  // wave-uniform

  const int l15 = lane & 15;
  const int khalf = (lane >> 4) << 1;

  v8f acc[2][2] = {};

#pragma unroll 4
  for (int k = 0; k < K; k += 4) {
    v2f a[2], b[2];
#pragma unroll
    for (int mi = 0; mi < 2; ++mi) {
      const int row = m_base + mi * 16 + l15;
      a[mi] = *(const v2f*)(A + (size_t)row * lda + k + khalf);
    }
#pragma unroll
    for (int ni = 0; ni < 2; ++ni) {
      const int col = n_base + ni * 16 + l15;
      if (TRANSB) {
        b[ni] = *(const v2f*)(Bm + (size_t)col * ldb + k + khalf);
      } else {
        b[ni].x = Bm[(size_t)(k + khalf) * ldb + col];
        b[ni].y = Bm[(size_t)(k + khalf + 1) * ldb + col];
      }
    }
    acc[0][0] = wmma_f32_4(a[0], b[0], acc[0][0]);
    acc[0][1] = wmma_f32_4(a[0], b[1], acc[0][1]);
    acc[1][0] = wmma_f32_4(a[1], b[0], acc[1][0]);
    acc[1][1] = wmma_f32_4(a[1], b[1], acc[1][1]);
  }

  const int rowadd = (lane >> 4) << 3;
  float lsum = 0.0f;
#pragma unroll
  for (int mi = 0; mi < 2; ++mi) {
#pragma unroll
    for (int ni = 0; ni < 2; ++ni) {
#pragma unroll
      for (int r = 0; r < 8; ++r) {
        const int row = m_base + mi * 16 + r + rowadd;
        const int col = n_base + ni * 16 + l15;
        float v = acc[mi][ni][r];
        if (EPI == 0 || EPI == 1 || EPI == 2) v += aux[col];
        if (EPI == 0) v = fmaxf(v, 0.0f);
        if (EPI == 2) {
          const float d = v - Xref[(size_t)row * ldc + col];
          lsum += d * d;
        } else if (EPI == 3) {
          const float d2 = aux[row] + aux[col] - 2.0f * v;
          C[(size_t)row * ldc + col] = (d2 > 0.0f) ? sqrtf(d2) : 0.0f;
        } else {
          C[(size_t)row * ldc + col] = v;
        }
      }
    }
  }
  if (EPI == 2) {
#pragma unroll
    for (int off = 16; off > 0; off >>= 1) lsum += __shfl_down(lsum, off, 32);
    if (lane == 0) atomicAdd(accum, lsum);
  }
}

// ---------------------------------------------------------------------------
// Split x into bf16 hi/lo planes with pure-integer RNE rounding.
// ---------------------------------------------------------------------------
__global__ __launch_bounds__(256) void decompose_bf16_kernel(
    const float* __restrict__ X, unsigned short* __restrict__ hi,
    unsigned short* __restrict__ lo, long long n) {
  const long long stride = (long long)gridDim.x * blockDim.x;
  for (long long i = (long long)blockIdx.x * blockDim.x + threadIdx.x; i < n;
       i += stride) {
    const float v = X[i];
    const unsigned u = __float_as_uint(v);
    const unsigned r = u + 0x7FFFu + ((u >> 16) & 1u);   // RNE to bf16
    const unsigned short h = (unsigned short)(r >> 16);
    const float res = v - __uint_as_float(((unsigned)h) << 16);
    const unsigned u2 = __float_as_uint(res);
    const unsigned r2 = u2 + 0x7FFFu + ((u2 >> 16) & 1u);
    hi[i] = h;
    lo[i] = (unsigned short)(r2 >> 16);
  }
}

// ---------------------------------------------------------------------------
// bf16x3 Gram: Dist = sqrt(relu(sq[i]+sq[j] - 2*(hi.hiT + hi.loT + lo.hiT))).
// Wave tile 64x32 (24 bf16 WMMAs per K=32 step, ~32 FLOP/B from cache),
// block = 8 waves (2x4) -> 128x128 tile.  A-frag layout (16-bit, ISA 7.12.2):
// lanes<16 K in {0..7,16..23} -> 16B loads at k+0,k+16 (lanes>=16: +8).
// B-frag (NT): same K layout as A -> 16B loads at k+0,k+16 (lanes>=16: +8).
// ---------------------------------------------------------------------------
__global__ __launch_bounds__(256) void gram_bf16x3_kernel(
    const unsigned short* __restrict__ Xhi,
    const unsigned short* __restrict__ Xlo, const float* __restrict__ sq,
    float* __restrict__ Dist, int N, int K) {
  const int lane = threadIdx.x & 31;
  const int wave = threadIdx.x >> 5;
  const int wm = wave & 1;   // 2 wave-rows  * 64
  const int wn = wave >> 1;  // 4 wave-cols  * 32
  const int m_base = blockIdx.y * 128 + wm * 64;
  const int n_base = blockIdx.x * 128 + wn * 32;
  const int l15 = lane & 15;
  const int koffA = (lane >> 4) << 3;   // 0 / 8  elements
  const int koffB = (lane >> 4) << 3;   // 0 / 8  elements

  v8f acc[4][2] = {};

#pragma unroll 2
  for (int k = 0; k < K; k += 32) {
    v16bf ahi[4], alo[4], bhi[2], blo[2];
#pragma unroll
    for (int mi = 0; mi < 4; ++mi) {
      const size_t base = (size_t)(m_base + mi * 16 + l15) * K + k + koffA;
      ahi[mi] = cat8(*(const v8bf*)(Xhi + base), *(const v8bf*)(Xhi + base + 16));
      alo[mi] = cat8(*(const v8bf*)(Xlo + base), *(const v8bf*)(Xlo + base + 16));
    }
#pragma unroll
    for (int ni = 0; ni < 2; ++ni) {
      const size_t base = (size_t)(n_base + ni * 16 + l15) * K + k + koffB;
      bhi[ni] = cat8(*(const v8bf*)(Xhi + base), *(const v8bf*)(Xhi + base + 16));
      blo[ni] = cat8(*(const v8bf*)(Xlo + base), *(const v8bf*)(Xlo + base + 16));
    }
    if (k + 128 < K) {  // uniform guard; per-lane L2 prefetch of next K-block
      __builtin_prefetch(Xhi + (size_t)(m_base + l15) * K + k + 128, 0, 0);
      __builtin_prefetch(Xhi + (size_t)(n_base + l15) * K + k + 128, 0, 0);
    }
#pragma unroll
    for (int mi = 0; mi < 4; ++mi) {
#pragma unroll
      for (int ni = 0; ni < 2; ++ni) {
        acc[mi][ni] = wmma_bf16_32(ahi[mi], bhi[ni], acc[mi][ni]);
        acc[mi][ni] = wmma_bf16_32(ahi[mi], blo[ni], acc[mi][ni]);
        acc[mi][ni] = wmma_bf16_32(alo[mi], bhi[ni], acc[mi][ni]);
      }
    }
  }

  const int rowadd = (lane >> 4) << 3;
#pragma unroll
  for (int mi = 0; mi < 4; ++mi) {
#pragma unroll
    for (int ni = 0; ni < 2; ++ni) {
#pragma unroll
      for (int r = 0; r < 8; ++r) {
        const int row = m_base + mi * 16 + r + rowadd;
        const int col = n_base + ni * 16 + l15;
        const float g = acc[mi][ni][r];
        const float d2 = sq[row] + sq[col] - 2.0f * g;
        Dist[(size_t)row * N + col] = (d2 > 0.0f) ? sqrtf(d2) : 0.0f;
      }
    }
  }
}

// ---------------------------------------------------------------------------
__global__ __launch_bounds__(256) void zero_ws_kernel(uint4* __restrict__ m16,
                                                      long long n16,
                                                      float* __restrict__ scal) {
  const long long stride = (long long)gridDim.x * blockDim.x;
  for (long long i = (long long)blockIdx.x * blockDim.x + threadIdx.x; i < n16;
       i += stride)
    m16[i] = make_uint4(0u, 0u, 0u, 0u);
  if (blockIdx.x == 0 && threadIdx.x < 16) scal[threadIdx.x] = 0.0f;
}

__global__ __launch_bounds__(256) void rownorm_kernel(const float* __restrict__ X,
                                                      float* __restrict__ out,
                                                      int D) {
  __shared__ float red[256];
  const int row = blockIdx.x;
  float s = 0.0f;
  for (int j = threadIdx.x; j < D; j += 256) {
    const float v = X[(size_t)row * D + j];
    s += v * v;
  }
  red[threadIdx.x] = s;
  __syncthreads();
  for (int off = 128; off > 0; off >>= 1) {
    if (threadIdx.x < off) red[threadIdx.x] += red[threadIdx.x + off];
    __syncthreads();
  }
  if (threadIdx.x == 0) out[row] = red[0];
}

// one block per row: 32KB row cached in LDS (320KB/WGP), 15 argmin sweeps,
// symmetric mask via two byte stores (benign same-value race)
__global__ __launch_bounds__(256) void knn_mask_kernel(
    const float* __restrict__ xdist, unsigned char* __restrict__ mask) {
  __shared__ float srow[BB];
  __shared__ float rv[256];
  __shared__ int ri[256];
  const int i = blockIdx.x;
  const float inf = __builtin_inff();
  for (int j = threadIdx.x; j < BB; j += 256)
    srow[j] = (j == i) ? inf : xdist[(size_t)i * BB + j];
  __syncthreads();
  for (int t = 0; t < KNN; ++t) {
    float lv = inf;
    int li = 0;
    for (int j = threadIdx.x; j < BB; j += 256) {
      const float v = srow[j];
      if (v < lv) { lv = v; li = j; }
    }
    rv[threadIdx.x] = lv;
    ri[threadIdx.x] = li;
    __syncthreads();
    for (int off = 128; off > 0; off >>= 1) {
      if (threadIdx.x < off && rv[threadIdx.x + off] < rv[threadIdx.x]) {
        rv[threadIdx.x] = rv[threadIdx.x + off];
        ri[threadIdx.x] = ri[threadIdx.x + off];
      }
      __syncthreads();
    }
    if (threadIdx.x == 0) {
      const int j = ri[0];
      srow[j] = inf;
      mask[(size_t)i * BB + j] = 1;
      mask[(size_t)j * BB + i] = 1;
    }
    __syncthreads();
  }
}

// scal: [0]=rec_sum [1]=xmax(bits) [2]=zmax(bits) [3]=cnt [4]=dist_sum
__global__ __launch_bounds__(256) void masked_reduce_kernel(
    const unsigned char* __restrict__ mask, const float* __restrict__ xd,
    const float* __restrict__ zd, float* __restrict__ scal, long long n) {
  __shared__ float rx[256], rz[256], rc[256];
  float xm = 0.0f, zm = 0.0f, c = 0.0f;
  const long long stride = (long long)gridDim.x * blockDim.x;
  for (long long i = (long long)blockIdx.x * blockDim.x + threadIdx.x; i < n;
       i += stride)
    if (mask[i]) {
      xm = fmaxf(xm, xd[i]);
      zm = fmaxf(zm, zd[i]);
      c += 1.0f;
    }
  rx[threadIdx.x] = xm; rz[threadIdx.x] = zm; rc[threadIdx.x] = c;
  __syncthreads();
  for (int off = 128; off > 0; off >>= 1) {
    if (threadIdx.x < off) {
      rx[threadIdx.x] = fmaxf(rx[threadIdx.x], rx[threadIdx.x + off]);
      rz[threadIdx.x] = fmaxf(rz[threadIdx.x], rz[threadIdx.x + off]);
      rc[threadIdx.x] += rc[threadIdx.x + off];
    }
    __syncthreads();
  }
  if (threadIdx.x == 0) {
    atomicMax((unsigned int*)&scal[1], __float_as_uint(rx[0]));  // vals >= 0
    atomicMax((unsigned int*)&scal[2], __float_as_uint(rz[0]));
    atomicAdd(&scal[3], rc[0]);
  }
}

__global__ __launch_bounds__(256) void masked_sum_kernel(
    const unsigned char* __restrict__ mask, const float* __restrict__ xd,
    const float* __restrict__ zd, float* __restrict__ scal, long long n) {
  __shared__ float rs[256];
  const float invx = 1.0f / (__uint_as_float(((const unsigned int*)scal)[1]) + EPSF);
  const float invz = 1.0f / (__uint_as_float(((const unsigned int*)scal)[2]) + EPSF);
  float s = 0.0f;
  const long long stride = (long long)gridDim.x * blockDim.x;
  for (long long i = (long long)blockIdx.x * blockDim.x + threadIdx.x; i < n;
       i += stride)
    if (mask[i]) {
      const float d = zd[i] * invz - xd[i] * invx;
      s += d * d;
    }
  rs[threadIdx.x] = s;
  __syncthreads();
  for (int off = 128; off > 0; off >>= 1) {
    if (threadIdx.x < off) rs[threadIdx.x] += rs[threadIdx.x + off];
    __syncthreads();
  }
  if (threadIdx.x == 0) atomicAdd(&scal[4], rs[0]);
}

__global__ void finalize_kernel(const float* __restrict__ scal,
                                float* __restrict__ out) {
  if (threadIdx.x == 0 && blockIdx.x == 0) {
    const float rec = scal[0] / ((float)BB * (float)DD);
    const float dist = scal[4] / scal[3];
    out[0] = rec + 1.0f * dist;  // LAM = 1.0
  }
}

// ---------------------------------------------------------------------------
// Workspace layout (bytes)
// ---------------------------------------------------------------------------
static const size_t OFF_XD   = 0;                              // 256MB
static const size_t OFF_ZD   = OFF_XD + (size_t)BB * BB * 4;   // 256MB
static const size_t OFF_MASK = OFF_ZD + (size_t)BB * BB * 4;   // 64MB
static const size_t OFF_H    = OFF_MASK + (size_t)BB * BB;
static const size_t OFF_H2   = OFF_H + (size_t)BB * HH * 4;
static const size_t OFF_Z    = OFF_H2 + (size_t)BB * HH * 4;
static const size_t OFF_XSQ  = OFF_Z + (size_t)BB * LL * 4;
static const size_t OFF_ZSQ  = OFF_XSQ + (size_t)BB * 4;
static const size_t OFF_SCAL = OFF_ZSQ + (size_t)BB * 4;
static const size_t OFF_XHI  = OFF_SCAL + 1024;                // 16MB bf16
static const size_t OFF_XLO  = OFF_XHI + (size_t)BB * DD * 2;  // 16MB bf16

extern "C" void kernel_launch(void* const* d_in, const int* in_sizes, int n_in,
                              void* d_out, int out_size, void* d_ws,
                              size_t ws_size, hipStream_t stream) {
  const float* x   = (const float*)d_in[0];
  const float* We1 = (const float*)d_in[1];
  const float* be1 = (const float*)d_in[2];
  const float* We2 = (const float*)d_in[3];
  const float* be2 = (const float*)d_in[4];
  const float* Wd1 = (const float*)d_in[5];
  const float* bd1 = (const float*)d_in[6];
  const float* Wd2 = (const float*)d_in[7];
  const float* bd2 = (const float*)d_in[8];

  char* ws = (char*)d_ws;
  float* xd          = (float*)(ws + OFF_XD);
  float* zd          = (float*)(ws + OFF_ZD);
  unsigned char* msk = (unsigned char*)(ws + OFF_MASK);
  float* h           = (float*)(ws + OFF_H);
  float* h2          = (float*)(ws + OFF_H2);
  float* z           = (float*)(ws + OFF_Z);
  float* xsq         = (float*)(ws + OFF_XSQ);
  float* zsq         = (float*)(ws + OFF_ZSQ);
  float* scal        = (float*)(ws + OFF_SCAL);
  unsigned short* xhi = (unsigned short*)(ws + OFF_XHI);
  unsigned short* xlo = (unsigned short*)(ws + OFF_XLO);

  const long long nBB = (long long)BB * BB;

  // 0) zero mask + scalar accumulators; split x into bf16 hi/lo planes
  zero_ws_kernel<<<4096, 256, 0, stream>>>((uint4*)msk, nBB / 16, scal);
  decompose_bf16_kernel<<<4096, 256, 0, stream>>>(x, xhi, xlo,
                                                  (long long)BB * DD);

  // 1) h = relu(x @ We1 + be1)
  gemm_wmma_f32<0, 0><<<dim3(HH / 128, BB / 64), 256, 0, stream>>>(
      x, We1, be1, nullptr, h, nullptr, BB, HH, DD, DD, HH, HH);
  // 2) z = h @ We2 + be2
  gemm_wmma_f32<1, 0><<<dim3(1, BB / 64), 256, 0, stream>>>(
      h, We2, be2, nullptr, z, nullptr, BB, LL, HH, HH, LL, LL);
  // 3) h2 = relu(z @ Wd1 + bd1)
  gemm_wmma_f32<0, 0><<<dim3(HH / 128, BB / 64), 256, 0, stream>>>(
      z, Wd1, bd1, nullptr, h2, nullptr, BB, HH, LL, LL, HH, HH);
  // 4) rec_sum += sum((h2@Wd2 + bd2 - x)^2)   fused MSE
  gemm_wmma_f32<2, 0><<<dim3(DD / 128, BB / 64), 256, 0, stream>>>(
      h2, Wd2, bd2, x, nullptr, &scal[0], BB, DD, HH, HH, DD, DD);

  // 5) row squared norms (f32, exact)
  rownorm_kernel<<<BB, 256, 0, stream>>>(x, xsq, DD);
  rownorm_kernel<<<BB, 256, 0, stream>>>(z, zsq, LL);

  // 6) x_dist via bf16x3 split-precision Gram (the 137 GFLOP op)
  gram_bf16x3_kernel<<<dim3(BB / 128, BB / 128), 256, 0, stream>>>(
      xhi, xlo, xsq, xd, BB, DD);
  // 7) z_dist via exact f32 WMMA (K=64, cheap)
  gemm_wmma_f32<3, 1><<<dim3(BB / 128, BB / 64), 256, 0, stream>>>(
      z, z, zsq, nullptr, zd, nullptr, BB, BB, LL, LL, LL, BB);

  // 8) per-row kNN (k=15) -> symmetric mask
  knn_mask_kernel<<<BB, 256, 0, stream>>>(xd, msk);
  // 9) masked max(x), max(z), count
  masked_reduce_kernel<<<4096, 256, 0, stream>>>(msk, xd, zd, scal, nBB);
  // 10) masked sum((z/zmax - x/xmax)^2)
  masked_sum_kernel<<<4096, 256, 0, stream>>>(msk, xd, zd, scal, nBB);
  // 11) out = rec/(B*D) + LAM * dist_sum/cnt
  finalize_kernel<<<1, 64, 0, stream>>>(scal, (float*)d_out);
}
